// Attention_62964220559726
// MI455X (gfx1250) — matrix operand order repaired and checked by reading, not compiled, not run on hardware
//
#include <hip/hip_runtime.h>

typedef unsigned short u16;
typedef __attribute__((ext_vector_type(16))) __bf16   v16bf;
typedef __attribute__((ext_vector_type(8)))  float    v8f;
typedef __attribute__((ext_vector_type(4)))  unsigned u32x4;
typedef __attribute__((address_space(3)))    u16      u16_as3;

union Frag { v16bf bf; u32x4 q[2]; };

#define BB   2
#define SS   2048
#define DD   4096
#define HH   32
#define KVHH 8
#define HDD  128
#define NREP 4
#define NQKV 6144   // H*HD + 2*KVH*HD

__device__ __forceinline__ u16 f2bf(float f) {
  unsigned u = __float_as_uint(f);
  unsigned r = u + 0x7fffu + ((u >> 16) & 1u);   // round-to-nearest-even
  return (u16)(r >> 16);
}

#define WMMA_BF16(A, B, C) \
  __builtin_amdgcn_wmma_f32_16x16x32_bf16(false, (A).bf, false, (B).bf, (short)0, (C), false, false)

// ---- CDNA5 async DMA: global -> LDS, tracked by ASYNCcnt ------------------
__device__ __forceinline__ void async_b128(const u16* g, const u16* lds) {
  unsigned lo = (unsigned)(size_t)(u16_as3*)lds;       // LDS byte offset (p3 is 32-bit on device)
  asm volatile("global_load_async_to_lds_b128 %0, %1, off"
               :: "v"(lo), "v"(g) : "memory");
}
__device__ __forceinline__ void wait_async0() {
#if __has_builtin(__builtin_amdgcn_s_wait_asynccnt)
  __builtin_amdgcn_s_wait_asynccnt(0);
#else
  asm volatile("s_wait_asynccnt 0x0" ::: "memory");
#endif
}

// ---------------------------------------------------------------------------
// Register-blocked GEMM with async-LDS double buffering.
// C[M x N](f32) = A[M x K](bf16) * BT[N x K]^T (bf16).
// Block = 128 threads = 4 waves (2x2) -> 128x128 C tile.
// Steady-state loop (last iteration peeled, no branches):
//   s_wait_asynccnt 0 -> barrier -> issue 8 async b128 DMAs (next panels)
//   -> ds_load_b128 fragments -> 16 v_wmma.
// All 4 waves share each 128x32 panel (no duplicate global traffic).
// ---------------------------------------------------------------------------
__global__ void __launch_bounds__(128) gemm_bf16_blk(const u16* __restrict__ A,
                                                     const u16* __restrict__ BT,
                                                     float* __restrict__ C,
                                                     int M, int N, int K) {
  __shared__ alignas(16) u16 At[2][128 * 32];   // 8 KB per buffer
  __shared__ alignas(16) u16 Bt[2][128 * 32];

  int t = threadIdx.x;
  int wave = t >> 5, lane = t & 31;
  int l = lane & 15, hi = lane >> 4;
  int wm = wave >> 1, wn = wave & 1;
  int nb = N >> 7;
  int mblk0 = (blockIdx.x / nb) * 128;
  int nblk0 = (blockIdx.x % nb) * 128;

  const u16* ag = A  + (size_t)(mblk0 + t) * K;   // thread t stages panel row t
  const u16* bg = BT + (size_t)(nblk0 + t) * K;

  v8f acc[4][4];
#pragma unroll
  for (int i = 0; i < 4; ++i)
#pragma unroll
    for (int j = 0; j < 4; ++j) acc[i][j] = {};

  // prologue: stage k=0 into buffer 0 (64 B/row => 4 b128 per panel per thread)
#pragma unroll
  for (int j = 0; j < 4; ++j) {
    async_b128(ag + j * 8, &At[0][t * 32 + j * 8]);
    async_b128(bg + j * 8, &Bt[0][t * 32 + j * 8]);
  }

  int buf = 0;
  for (int k = 0; k < K - 32; k += 32) {   // steady state: always stages next
    wait_async0();        // our DMAs for 'buf' complete
    __syncthreads();      // all waves' DMAs done; prior readers of buf^1 done
#pragma unroll
    for (int j = 0; j < 4; ++j) {
      async_b128(ag + k + 32 + j * 8, &At[buf ^ 1][t * 32 + j * 8]);
      async_b128(bg + k + 32 + j * 8, &Bt[buf ^ 1][t * 32 + j * 8]);
    }
    Frag fa[4], fb[4];
#pragma unroll
    for (int i = 0; i < 4; ++i) {
      const u16* ar = &At[buf][(wm * 64 + i * 16 + l) * 32];
      const u16* br = &Bt[buf][(wn * 64 + i * 16 + l) * 32];
      fa[i].q[0] = *(const u32x4*)(ar + hi * 8);
      fa[i].q[1] = *(const u32x4*)(ar + 16 + hi * 8);
      fb[i].q[0] = *(const u32x4*)(br + hi * 8);
      fb[i].q[1] = *(const u32x4*)(br + 16 + hi * 8);
    }
#pragma unroll
    for (int i = 0; i < 4; ++i)
#pragma unroll
      for (int j = 0; j < 4; ++j) acc[i][j] = WMMA_BF16(fa[i], fb[j], acc[i][j]);
    buf ^= 1;
  }

  // epilogue: last K-tile, nothing left to stage
  wait_async0();
  __syncthreads();
  {
    Frag fa[4], fb[4];
#pragma unroll
    for (int i = 0; i < 4; ++i) {
      const u16* ar = &At[buf][(wm * 64 + i * 16 + l) * 32];
      const u16* br = &Bt[buf][(wn * 64 + i * 16 + l) * 32];
      fa[i].q[0] = *(const u32x4*)(ar + hi * 8);
      fa[i].q[1] = *(const u32x4*)(ar + 16 + hi * 8);
      fb[i].q[0] = *(const u32x4*)(br + hi * 8);
      fb[i].q[1] = *(const u32x4*)(br + 16 + hi * 8);
    }
#pragma unroll
    for (int i = 0; i < 4; ++i)
#pragma unroll
      for (int j = 0; j < 4; ++j) acc[i][j] = WMMA_BF16(fa[i], fb[j], acc[i][j]);
  }

  int mblk = mblk0 + wm * 64, nblk = nblk0 + wn * 64;
#pragma unroll
  for (int i = 0; i < 4; ++i)
#pragma unroll
    for (int j = 0; j < 4; ++j) {
      float* cp = C + (size_t)(mblk + i * 16 + hi * 8) * N + nblk + j * 16 + l;
#pragma unroll
      for (int r = 0; r < 8; ++r) cp[(size_t)r * N] = acc[i][j][r];
    }
}

// --------------------------- conversions -----------------------------------
__global__ void cvt_bf16(const float* __restrict__ in, u16* __restrict__ out, int n) {
  int i = blockIdx.x * blockDim.x + threadIdx.x;
  if (i < n) out[i] = f2bf(in[i]);
}

// in: R x C (row-major f32) -> out: C x R (bf16, transposed)
__global__ void cvt_bf16_T(const float* __restrict__ in, u16* __restrict__ out,
                           int R, int C) {
  int i = blockIdx.x * blockDim.x + threadIdx.x;
  if (i >= R * C) return;
  int c = i % C, r = i / C;
  out[(size_t)c * R + r] = f2bf(in[i]);
}

// --------------------------- RoPE + layout prep -----------------------------
// XOUT rows = b*S+s with stride NQKV: cols [0,4096)=Q, [4096,5120)=K, [5120,6144)=V
__global__ void rope_q(const float* __restrict__ xo, const float* __restrict__ cs,
                       const float* __restrict__ sn, u16* __restrict__ qb) {
  int i = blockIdx.x * blockDim.x + threadIdx.x;
  if (i >= BB * SS * HH * (HDD / 2)) return;
  int p = i & 63; int t = i >> 6;
  int h = t % HH; t /= HH;
  int s = t % SS; int b = t / SS;
  size_t ii = ((size_t)b * SS + s) * NQKV + h * HDD + 2 * p;
  float re = xo[ii], im = xo[ii + 1];
  float c = cs[s * 64 + p], si = sn[s * 64 + p];
  size_t io = (((size_t)b * HH + h) * SS + s) * HDD + 2 * p;  // (b,h,s,d)
  qb[io]     = f2bf(re * c - im * si);
  qb[io + 1] = f2bf(re * si + im * c);
}

__global__ void rope_k(const float* __restrict__ xo, const float* __restrict__ cs,
                       const float* __restrict__ sn, u16* __restrict__ kb) {
  int i = blockIdx.x * blockDim.x + threadIdx.x;
  if (i >= BB * SS * KVHH * (HDD / 2)) return;
  int p = i & 63; int t = i >> 6;
  int kv = t % KVHH; t /= KVHH;
  int s = t % SS; int b = t / SS;
  size_t ii = ((size_t)b * SS + s) * NQKV + (HH * HDD) + kv * HDD + 2 * p;
  float re = xo[ii], im = xo[ii + 1];
  float c = cs[s * 64 + p], si = sn[s * 64 + p];
  size_t io = (((size_t)b * KVHH + kv) * SS + s) * HDD + 2 * p;  // (b,kvh,s,d)
  kb[io]     = f2bf(re * c - im * si);
  kb[io + 1] = f2bf(re * si + im * c);
}

__global__ void vtrans(const float* __restrict__ xo, u16* __restrict__ vt) {
  int i = blockIdx.x * blockDim.x + threadIdx.x;
  if (i >= BB * SS * KVHH * HDD) return;
  int d = i & 127; int t = i >> 7;
  int kv = t % KVHH; t /= KVHH;
  int s = t % SS; int b = t / SS;
  size_t ii = ((size_t)b * SS + s) * NQKV + (HH * HDD + KVHH * HDD) + kv * HDD + d;
  vt[(((size_t)b * KVHH + kv) * HDD + d) * SS + s] = f2bf(xo[ii]);  // (b,kvh,d,s)
}

// ---------------------------------------------------------------------------
// Flash attention, 32 q-rows per wave (two 16-row groups share K/V fragments).
// Per 32-key block: 16 K-loads + 16 V-loads feed 32 v_wmma. Online softmax in
// f32; P transposed C-layout -> A-layout through LDS (ds_store/ds_load_b128).
// ---------------------------------------------------------------------------
__global__ void __launch_bounds__(32) flash_attn(const u16* __restrict__ Qb,
                                                 const u16* __restrict__ Kb,
                                                 const u16* __restrict__ Vt,
                                                 u16* __restrict__ Ctx) {
  const int qtiles = SS / 32;
  int qt = blockIdx.x % qtiles;
  int bh = blockIdx.x / qtiles;
  int h = bh % HH, b = bh / HH;
  int kvh = h / NREP;
  int lane = threadIdx.x;
  int l = lane & 15, hi = lane >> 4;
  int qbase = qt * 32;

  const u16* qp = Qb + ((size_t)(b * HH + h) * SS + qbase) * HDD;
  const u16* kp = Kb + (size_t)(b * KVHH + kvh) * SS * HDD;
  const u16* vp = Vt + (size_t)(b * KVHH + kvh) * HDD * SS;

  Frag fq[2][4];
#pragma unroll
  for (int g = 0; g < 2; ++g) {
    const u16* qrow = qp + (size_t)(g * 16 + l) * HDD;
#pragma unroll
    for (int c = 0; c < 4; ++c) {
      fq[g][c].q[0] = *(const u32x4*)(qrow + c * 32 + hi * 8);
      fq[g][c].q[1] = *(const u32x4*)(qrow + c * 32 + 16 + hi * 8);
    }
  }

  v8f O[2][8];
#pragma unroll
  for (int g = 0; g < 2; ++g)
#pragma unroll
    for (int c = 0; c < 8; ++c) O[g][c] = {};
  float mrow[2][8], lrow[2][8];
#pragma unroll
  for (int g = 0; g < 2; ++g)
#pragma unroll
    for (int r = 0; r < 8; ++r) { mrow[g][r] = -3.0e38f; lrow[g][r] = 0.0f; }

  __shared__ alignas(16) u16 pls[2][16 * 32];
  const float scale = 0.08838834764831845f;  // 1/sqrt(128)

  int kend = qbase + 32;
  for (int kb = 0; kb < kend; kb += 32) {
    v8f sc[2][2];
#pragma unroll
    for (int g = 0; g < 2; ++g) { sc[g][0] = {}; sc[g][1] = {}; }

    const u16* krow0 = kp + (size_t)(kb + l) * HDD;
    const u16* krow1 = kp + (size_t)(kb + 16 + l) * HDD;
#pragma unroll
    for (int c = 0; c < 4; ++c) {
      Frag fb0, fb1;
      fb0.q[0] = *(const u32x4*)(krow0 + c * 32 + hi * 8);
      fb0.q[1] = *(const u32x4*)(krow0 + c * 32 + 16 + hi * 8);
      fb1.q[0] = *(const u32x4*)(krow1 + c * 32 + hi * 8);
      fb1.q[1] = *(const u32x4*)(krow1 + c * 32 + 16 + hi * 8);
      sc[0][0] = WMMA_BF16(fq[0][c], fb0, sc[0][0]);
      sc[1][0] = WMMA_BF16(fq[1][c], fb0, sc[1][0]);
      sc[0][1] = WMMA_BF16(fq[0][c], fb1, sc[0][1]);
      sc[1][1] = WMMA_BF16(fq[1][c], fb1, sc[1][1]);
    }

    // ---- scale + causal mask + online softmax per row group ----
#pragma unroll
    for (int g = 0; g < 2; ++g) {
#pragma unroll
      for (int r = 0; r < 8; ++r) {
        int q  = qbase + g * 16 + r + 8 * hi;
        int k0 = kb + l, k1 = kb + 16 + l;
        float v0 = (k0 <= q) ? sc[g][0][r] * scale : -1.0e9f;
        float v1 = (k1 <= q) ? sc[g][1][r] * scale : -1.0e9f;
        float lm = fmaxf(v0, v1);
#pragma unroll
        for (int m = 1; m < 16; m <<= 1) lm = fmaxf(lm, __shfl_xor(lm, m, 32));
        float mnew  = fmaxf(mrow[g][r], lm);
        float alpha = __expf(mrow[g][r] - mnew);
        float p0 = __expf(v0 - mnew);
        float p1 = __expf(v1 - mnew);
        float ps = p0 + p1;
#pragma unroll
        for (int m = 1; m < 16; m <<= 1) ps += __shfl_xor(ps, m, 32);
        lrow[g][r] = lrow[g][r] * alpha + ps;
        mrow[g][r] = mnew;
#pragma unroll
        for (int c = 0; c < 8; ++c) O[g][c][r] *= alpha;
        int row = r + 8 * hi;
        pls[g][row * 32 + l]      = f2bf(p0);
        pls[g][row * 32 + 16 + l] = f2bf(p1);
      }
    }
    __syncthreads();

    Frag fp[2];
#pragma unroll
    for (int g = 0; g < 2; ++g) {
      fp[g].q[0] = *(const u32x4*)(&pls[g][l * 32 + hi * 8]);
      fp[g].q[1] = *(const u32x4*)(&pls[g][l * 32 + 16 + hi * 8]);
    }
#pragma unroll
    for (int c = 0; c < 8; ++c) {     // O(32x128) += P(32x32) * V(32x128)
      Frag fbv;
      const u16* vrow = vp + (size_t)(c * 16 + l) * SS + kb + hi * 8;
      fbv.q[0] = *(const u32x4*)(vrow);
      fbv.q[1] = *(const u32x4*)(vrow + 16);
      O[0][c] = WMMA_BF16(fp[0], fbv, O[0][c]);
      O[1][c] = WMMA_BF16(fp[1], fbv, O[1][c]);
    }
  }

  // ctx as (b,s,h,d) bf16 => rows b*S+s, cols h*HD+d for the final GEMM
#pragma unroll
  for (int g = 0; g < 2; ++g)
#pragma unroll
    for (int c = 0; c < 8; ++c)
#pragma unroll
      for (int r = 0; r < 8; ++r) {
        int q = qbase + g * 16 + r + 8 * hi;
        int d = c * 16 + l;
        Ctx[(((size_t)b * SS + q) * HH + h) * HDD + d] = f2bf(O[g][c][r] / lrow[g][r]);
      }
}

// ---------------------------------------------------------------------------
extern "C" void kernel_launch(void* const* d_in, const int* in_sizes, int n_in,
                              void* d_out, int out_size, void* d_ws, size_t ws_size,
                              hipStream_t stream) {
  (void)in_sizes; (void)n_in; (void)out_size; (void)ws_size;
  const float* x  = (const float*)d_in[0];
  const float* fc = (const float*)d_in[3];
  const float* fs = (const float*)d_in[4];
  const float* wq = (const float*)d_in[6];
  const float* wk = (const float*)d_in[7];
  const float* wv = (const float*)d_in[8];
  const float* wo = (const float*)d_in[9];
  float* out = (float*)d_out;

  const size_t M = (size_t)BB * SS;  // 4096 token rows
  char* ws = (char*)d_ws;
  size_t off = 0;
  u16*   XB    = (u16*)(ws + off); off += M * DD * 2;
  u16*   WQKVT = (u16*)(ws + off); off += (size_t)NQKV * DD * 2;  // Q|K|V transposed, contiguous
  u16*   WOT   = (u16*)(ws + off); off += (size_t)DD * DD * 2;
  float* XOUT  = (float*)(ws + off); off += M * NQKV * 4;         // fused QKV projection
  u16*   QB    = (u16*)(ws + off); off += M * DD * 2;
  u16*   KB    = (u16*)(ws + off); off += M * 1024 * 2;
  u16*   VT    = (u16*)(ws + off); off += M * 1024 * 2;
  u16*   CTX   = (u16*)(ws + off); off += M * DD * 2;

  const int tb = 256;
  cvt_bf16  <<<(int)((M * DD + tb - 1) / tb), tb, 0, stream>>>(x, XB, (int)(M * DD));
  cvt_bf16_T<<<(int)(((size_t)DD * DD   + tb - 1) / tb), tb, 0, stream>>>(wq, WQKVT, DD, DD);
  cvt_bf16_T<<<(int)(((size_t)DD * 1024 + tb - 1) / tb), tb, 0, stream>>>(wk, WQKVT + (size_t)4096 * DD, DD, 1024);
  cvt_bf16_T<<<(int)(((size_t)DD * 1024 + tb - 1) / tb), tb, 0, stream>>>(wv, WQKVT + (size_t)5120 * DD, DD, 1024);
  cvt_bf16_T<<<(int)(((size_t)DD * DD   + tb - 1) / tb), tb, 0, stream>>>(wo, WOT, DD, DD);

  // Fused QKV projection: (4096 x 4096) x (4096 x 6144) -> 4096 x 6144
  gemm_bf16_blk<<<(4096 / 128) * (NQKV / 128), 128, 0, stream>>>(XB, WQKVT, XOUT, 4096, NQKV, 4096);

  rope_q<<<(BB * SS * HH * 64) / tb, tb, 0, stream>>>(XOUT, fc, fs, QB);
  rope_k<<<(BB * SS * KVHH * 64) / tb, tb, 0, stream>>>(XOUT, fc, fs, KB);
  vtrans<<<(BB * SS * KVHH * HDD) / tb, tb, 0, stream>>>(XOUT, VT);

  flash_attn<<<BB * HH * (SS / 32), 32, 0, stream>>>(QB, KB, VT, CTX);

  // Output projection: (4096 x 4096) x (4096 x 4096) -> d_out
  gemm_bf16_blk<<<(4096 / 128) * (4096 / 128), 128, 0, stream>>>(CTX, WOT, out, 4096, 4096, 4096);
}
